// STDPClassifyLayer_11132555231658
// MI455X (gfx1250) — matrix operand order repaired and checked by reading, not compiled, not run on hardware
//
#include <hip/hip_runtime.h>

// ---------------------------------------------------------------------------
// LIF + STDP scan for MI455X (gfx1250).
//  kernel 0: init masters/f16 shadows, zero state, zero grid-barrier counter
//  kernel 1: i_seq = x @ Wi^T ; A-panel staged in LDS (f32->f16 once), then
//            v_wmma_f32_16x16x32_f16 with 4 N-tiles per A-fragment.
//  kernel 2: persistent 16-WG scan; 4 phases/step, grid barrier + cluster
//            barrier between phases; all GEMMs via f16 WMMA, f32 accumulate.
// ---------------------------------------------------------------------------

typedef __attribute__((ext_vector_type(16))) _Float16 v16h;
typedef __attribute__((ext_vector_type(8)))  _Float16 v8h;
typedef __attribute__((ext_vector_type(8)))  float    v8f;

#define NB   64
#define NT   500
#define NIN  512
#define NHID 1024
#define NOUT 256

constexpr float DECAY  = 0.9512294245007140f;   // exp(-1/20)
constexpr float OMD    = 1.0f - DECAY;          // (1 - decay_trace)
constexpr float LRB    = 1.0e-3f / 64.0f;       // LR / B

struct WS {
  unsigned* bar;
  float*    iseq;                 // [T][B][IN] f32
  _Float16* wif16;                // [IN][IN]
  float*    whf32; _Float16* whf16;   // [HID][IN]
  float*    wof32; _Float16* wof16;   // [OUT][HID]
  float *v1, *z1, *tr1;           // [B][IN]
  float *v2, *z2, *tr2;           // [B][HID]
  float *v3, *z3, *tr3;           // [B][OUT]
  _Float16 *z1h, *tr1t;           // [B][IN], [IN][B]
  _Float16 *z2h, *tr2t;           // [B][HID], [HID][B]
  _Float16 *tr3t;                 // [OUT][B]
};

// --- WMMA tile loaders (contiguous k order k = 16*(lane/16) + s, same for A and B) -----------------
// A (16x32 f16, row-major memory [M][K]): lane<16 holds row M=lane,
// K=k0..k0+15; lane>=16 holds K shifted by +16.
__device__ __forceinline__ v16h load_a16(const _Float16* base, int m0, int k0,
                                         int stride, int lane) {
  const int row = m0 + (lane & 15);
  const int kb  = k0 + ((lane >> 4) << 4);
  const _Float16* p = base + (size_t)row * stride + kb;
  v8h lo = *(const v8h*)p;
  v8h hi = *(const v8h*)(p + 8);
  v16h r;
#pragma unroll
  for (int i = 0; i < 8; ++i) { r[i] = lo[i]; r[i + 8] = hi[i]; }
  return r;
}

// B (32x16 f16). B[k][n]; memory laid out as B^T row-major [N][K]:
// lane<16 -> column N=lane, K=k0..k0+15 (contiguous 32B); lane>=16 -> K+16.
__device__ __forceinline__ v16h load_b16(const _Float16* base, int n0, int k0,
                                         int stride, int lane) {
  const int row = n0 + (lane & 15);
  const int kb  = k0 + ((lane >> 4) << 4);
  return *(const v16h*)(base + (size_t)row * stride + kb);
}

__device__ __forceinline__ v8f wmma16(v16h a, v16h b, v8f c) {
  return __builtin_amdgcn_wmma_f32_16x16x32_f16(false, a, false, b,
                                                (short)0, c, false, false);
}

// Monotonic atomic grid barrier (correct for any dispatch of 16 co-resident
// WGs) + s_cluster_barrier (CDNA5 HW fast path when launched as a cluster;
// architectural NOP when ClusterID==0).
__device__ __forceinline__ void grid_barrier(unsigned* bar, unsigned& phase) {
  __threadfence();
  __syncthreads();
  if (threadIdx.x == 0) {
    ++phase;
    atomicAdd(bar, 1u);
    const unsigned target = phase * gridDim.x;
    while (atomicAdd(bar, 0u) < target) { __builtin_amdgcn_s_sleep(8); }
  }
  __builtin_amdgcn_s_cluster_barrier();
  __syncthreads();
  __threadfence();
}

// ---------------------------------------------------------------------------
__global__ void k_init(WS w, const float* wi, const float* wh, const float* wo) {
  const int g  = blockIdx.x * blockDim.x + threadIdx.x;
  const int gs = gridDim.x * blockDim.x;
  for (int i = g; i < NIN * NIN;  i += gs) w.wif16[i] = (_Float16)wi[i];
  for (int i = g; i < NHID * NIN; i += gs) {
    const float v = wh[i]; w.whf32[i] = v; w.whf16[i] = (_Float16)v;
  }
  for (int i = g; i < NOUT * NHID; i += gs) {
    const float v = wo[i]; w.wof32[i] = v; w.wof16[i] = (_Float16)v;
  }
  for (int i = g; i < NB * NIN;  i += gs) { w.v1[i] = 0.f; w.z1[i] = 0.f; w.tr1[i] = 0.f; }
  for (int i = g; i < NB * NHID; i += gs) { w.v2[i] = 0.f; w.z2[i] = 0.f; w.tr2[i] = 0.f; }
  for (int i = g; i < NB * NOUT; i += gs) { w.v3[i] = 0.f; w.z3[i] = 0.f; w.tr3[i] = 0.f; }
  if (g == 0) *w.bar = 0u;
}

// i_seq[t][b][:] = x[b][t][:] @ Wi^T   (rows r = b*T + t, M = B*T = 32000)
// A-panel (16 rows x 512 k, f32 spikes) converted to f16 in LDS once/block.
__global__ void __launch_bounds__(256) k_iseq(WS w, const float* x) {
  __shared__ __align__(32) _Float16 sA[16 * NIN];     // 16 KB of WGP LDS
  const int tid  = threadIdx.x;
  const int lane = tid & 31;
  const int hi   = lane >> 4;
  const int wv   = tid >> 5;                          // 0..7
  const int r0   = blockIdx.x * 16;

  // Stage: 8192 contiguous floats (full 16 rows) -> f16 LDS, 32 per thread.
  {
    const int base = tid * 32;
    const float* src = x + (size_t)r0 * NIN + base;
#pragma unroll
    for (int c = 0; c < 32; c += 8) {
      const float4 f0 = *(const float4*)(src + c);
      const float4 f1 = *(const float4*)(src + c + 4);
      v8h h;
      h[0] = (_Float16)f0.x; h[1] = (_Float16)f0.y;
      h[2] = (_Float16)f0.z; h[3] = (_Float16)f0.w;
      h[4] = (_Float16)f1.x; h[5] = (_Float16)f1.y;
      h[6] = (_Float16)f1.z; h[7] = (_Float16)f1.w;
      *(v8h*)&sA[base + c] = h;
    }
  }
  __syncthreads();

  // Each wave: 4 N-tiles {wv, wv+8, wv+16, wv+24}; one A fragment feeds 4 WMMA.
  v8f acc[4] = {{}, {}, {}, {}};
  const int row = lane & 15;
  for (int kt = 0; kt < NIN / 32; ++kt) {
    const int kb = kt * 32 + (hi << 4);
    const v8h lo = *(const v8h*)&sA[row * NIN + kb];
    const v8h hh = *(const v8h*)&sA[row * NIN + kb + 8];
    v16h a;
#pragma unroll
    for (int i = 0; i < 8; ++i) { a[i] = lo[i]; a[i + 8] = hh[i]; }
#pragma unroll
    for (int u = 0; u < 4; ++u) {
      const v16h b = load_b16(w.wif16, (wv + 8 * u) * 16, kt * 32, NIN, lane);
      acc[u] = wmma16(a, b, acc[u]);
    }
  }
#pragma unroll
  for (int u = 0; u < 4; ++u) {
    const int col = (wv + 8 * u) * 16 + (lane & 15);
#pragma unroll
    for (int j = 0; j < 8; ++j) {
      const int r  = r0 + j + (hi << 3);
      const int t  = r % NT;
      const int bb = r / NT;
      w.iseq[((size_t)t * NB + bb) * NIN + col] = acc[u][j];
    }
  }
}

// Persistent scan: 16 WGs x 256 threads = 128 waves.
__global__ void __launch_bounds__(256) k_scan(WS w, float* out1, float* out2,
                                              float* out3, const int* trainp) {
  const int tid  = threadIdx.x;
  const int g    = blockIdx.x * 256 + tid;
  const int lane = tid & 31;
  const int hi   = lane >> 4;
  const int wid  = g >> 5;                  // 0..127
  const int train = *trainp;
  unsigned phase = 0;

  for (int t = 0; t < NT; ++t) {
    // ---- Phase A: layer-1 LIF + trace; 8 contiguous elems/thread ----------
    {
      const int base = g * 8;               // 4096 threads * 8 == 32768 exactly
      const int bb = base >> 9;
      const int i0 = base & (NIN - 1);
      float inp[8], vv[8], zz[8], to[8];
      *(float4*)&inp[0] = *(const float4*)&w.iseq[((size_t)t * NB + bb) * NIN + i0];
      *(float4*)&inp[4] = *(const float4*)&w.iseq[((size_t)t * NB + bb) * NIN + i0 + 4];
      *(float4*)&vv[0]  = *(const float4*)&w.v1[base];
      *(float4*)&vv[4]  = *(const float4*)&w.v1[base + 4];
      *(float4*)&zz[0]  = *(const float4*)&w.z1[base];
      *(float4*)&zz[4]  = *(const float4*)&w.z1[base + 4];
      v8h zh;
#pragma unroll
      for (int j = 0; j < 8; ++j) {
        const float v = DECAY * vv[j] * (1.0f - zz[j]) + inp[j];
        const float zn = (v > 1.0f) ? 1.0f : 0.0f;
        vv[j] = v; to[j] = zn; zh[j] = (_Float16)zn;
      }
      *(float4*)&w.v1[base]     = *(float4*)&vv[0];
      *(float4*)&w.v1[base + 4] = *(float4*)&vv[4];
      *(float4*)&w.z1[base]     = *(float4*)&to[0];
      *(float4*)&w.z1[base + 4] = *(float4*)&to[4];
      *(v8h*)&w.z1h[base] = zh;
      float* o1 = &out1[((size_t)bb * NT + t) * NIN + i0];
      *(float4*)o1       = *(float4*)&to[0];
      *(float4*)(o1 + 4) = *(float4*)&to[4];
      if (train) {
        float tr[8];
        *(float4*)&tr[0] = *(const float4*)&w.tr1[base];
        *(float4*)&tr[4] = *(const float4*)&w.tr1[base + 4];
#pragma unroll
        for (int j = 0; j < 8; ++j) {
          tr[j] = DECAY * tr[j] + OMD * to[j];
          w.tr1t[(i0 + j) * NB + bb] = (_Float16)tr[j];
        }
        *(float4*)&w.tr1[base]     = *(float4*)&tr[0];
        *(float4*)&w.tr1[base + 4] = *(float4*)&tr[4];
      }
    }
    grid_barrier(w.bar, phase);

    // ---- Phase B: z2 = LIF(0.2 * z1 @ Wh^T), 256 tiles over 128 waves -----
    for (int q = wid; q < (NB / 16) * (NHID / 16); q += 128) {
      const int mt = q & 3, nt = q >> 2;
      const _Float16* brow = w.whf16 + (size_t)(nt * 16 + (lane & 15)) * NIN
                                     + ((lane >> 4) << 4);
      v8f acc = {};
      for (int kt = 0; kt < NIN / 32; ++kt) {
        if (kt + 2 < NIN / 32) __builtin_prefetch(brow + (kt + 2) * 32, 0, 0);
        const v16h a = load_a16(w.z1h, mt * 16, kt * 32, NIN, lane);
        const v16h b = *(const v16h*)(brow + kt * 32);
        acc = wmma16(a, b, acc);
      }
#pragma unroll
      for (int j = 0; j < 8; ++j) {
        const int bb  = mt * 16 + j + (hi << 3);
        const int h   = nt * 16 + (lane & 15);
        const int idx = bb * NHID + h;
        float v = w.v2[idx];
        const float z = w.z2[idx];
        v = DECAY * v * (1.0f - z) + 0.2f * acc[j];
        const float zn = (v > 1.0f) ? 1.0f : 0.0f;
        w.v2[idx] = v; w.z2[idx] = zn; w.z2h[idx] = (_Float16)zn;
        out2[((size_t)bb * NT + t) * NHID + h] = zn;
        if (train) {
          const float tr = DECAY * w.tr2[idx] + OMD * zn;
          w.tr2[idx] = tr;
          w.tr2t[h * NB + bb] = (_Float16)tr;
        }
      }
    }
    grid_barrier(w.bar, phase);

    // ---- Phase C (WG 0..11): Wh += LR/B * tr2^T tr1  (2048 tiles, K=64) ---
    // ---- Phase D (WG 12..15): z3 = LIF(0.2 * z2 @ Wo^T)  (64 tiles) -------
    if (blockIdx.x < 12) {
      if (train) {
        for (int q = wid; q < (NHID / 16) * (NIN / 16); q += 96) {
          const int mt = q >> 5, nt = q & 31;
          v8f acc = {};
#pragma unroll
          for (int kt = 0; kt < 2; ++kt) {
            const v16h a = load_a16(w.tr2t, mt * 16, kt * 32, NB, lane);
            const v16h b = load_b16(w.tr1t, nt * 16, kt * 32, NB, lane);
            acc = wmma16(a, b, acc);
          }
#pragma unroll
          for (int j = 0; j < 8; ++j) {
            const int h   = mt * 16 + j + (hi << 3);
            const int i   = nt * 16 + (lane & 15);
            const int idx = h * NIN + i;
            const float wv = w.whf32[idx] + LRB * acc[j];
            w.whf32[idx] = wv;
            w.whf16[idx] = (_Float16)wv;
          }
        }
      }
    } else {
      const int dw = wid - 96;                  // 0..31
      for (int q = dw; q < (NB / 16) * (NOUT / 16); q += 32) {
        const int mt = q & 3, nt = q >> 2;
        const _Float16* brow = w.wof16 + (size_t)(nt * 16 + (lane & 15)) * NHID
                                       + ((lane >> 4) << 4);
        v8f acc = {};
        for (int kt = 0; kt < NHID / 32; ++kt) {
          if (kt + 2 < NHID / 32) __builtin_prefetch(brow + (kt + 2) * 32, 0, 0);
          const v16h a = load_a16(w.z2h, mt * 16, kt * 32, NHID, lane);
          const v16h b = *(const v16h*)(brow + kt * 32);
          acc = wmma16(a, b, acc);
        }
#pragma unroll
        for (int j = 0; j < 8; ++j) {
          const int bb  = mt * 16 + j + (hi << 3);
          const int o   = nt * 16 + (lane & 15);
          const int idx = bb * NOUT + o;
          float v = w.v3[idx];
          const float z = w.z3[idx];
          v = DECAY * v * (1.0f - z) + 0.2f * acc[j];
          const float zn = (v > 1.0f) ? 1.0f : 0.0f;
          w.v3[idx] = v; w.z3[idx] = zn;
          out3[((size_t)bb * NT + t) * NOUT + o] = zn;
          if (train) {
            const float tr = DECAY * w.tr3[idx] + OMD * zn;
            w.tr3[idx] = tr;
            w.tr3t[o * NB + bb] = (_Float16)tr;
          }
        }
      }
    }
    grid_barrier(w.bar, phase);

    // ---- Phase E: Wo += LR/B * tr3^T tr2 (1024 tiles over 128 waves) ------
    if (train) {
      for (int q = wid; q < (NOUT / 16) * (NHID / 16); q += 128) {
        const int mt = q >> 6, nt = q & 63;
        v8f acc = {};
#pragma unroll
        for (int kt = 0; kt < 2; ++kt) {
          const v16h a = load_a16(w.tr3t, mt * 16, kt * 32, NB, lane);
          const v16h b = load_b16(w.tr2t, nt * 16, kt * 32, NB, lane);
          acc = wmma16(a, b, acc);
        }
#pragma unroll
        for (int j = 0; j < 8; ++j) {
          const int o   = mt * 16 + j + (hi << 3);
          const int h   = nt * 16 + (lane & 15);
          const int idx = o * NHID + h;
          const float wv = w.wof32[idx] + LRB * acc[j];
          w.wof32[idx] = wv;
          w.wof16[idx] = (_Float16)wv;
        }
      }
    }
    grid_barrier(w.bar, phase);
  }
}

// ---------------------------------------------------------------------------
extern "C" void kernel_launch(void* const* d_in, const int* in_sizes, int n_in,
                              void* d_out, int out_size, void* d_ws, size_t ws_size,
                              hipStream_t stream) {
  (void)in_sizes; (void)n_in; (void)out_size; (void)ws_size;
  const float* x  = (const float*)d_in[0];
  const float* wi = (const float*)d_in[1];
  const float* wh = (const float*)d_in[2];
  const float* wo = (const float*)d_in[3];
  const int* train = (const int*)d_in[4];

  char* p = (char*)d_ws;
  auto take = [&](size_t bytes) -> char* {
    char* r = p; p += (bytes + 255) & ~(size_t)255; return r;
  };

  WS w;
  w.bar   = (unsigned*)take(256);
  w.iseq  = (float*)take((size_t)NT * NB * NIN * 4);
  w.wif16 = (_Float16*)take((size_t)NIN * NIN * 2);
  w.whf32 = (float*)take((size_t)NHID * NIN * 4);
  w.whf16 = (_Float16*)take((size_t)NHID * NIN * 2);
  w.wof32 = (float*)take((size_t)NOUT * NHID * 4);
  w.wof16 = (_Float16*)take((size_t)NOUT * NHID * 2);
  w.v1  = (float*)take((size_t)NB * NIN * 4);
  w.z1  = (float*)take((size_t)NB * NIN * 4);
  w.tr1 = (float*)take((size_t)NB * NIN * 4);
  w.v2  = (float*)take((size_t)NB * NHID * 4);
  w.z2  = (float*)take((size_t)NB * NHID * 4);
  w.tr2 = (float*)take((size_t)NB * NHID * 4);
  w.v3  = (float*)take((size_t)NB * NOUT * 4);
  w.z3  = (float*)take((size_t)NB * NOUT * 4);
  w.tr3 = (float*)take((size_t)NB * NOUT * 4);
  w.z1h  = (_Float16*)take((size_t)NB * NIN * 2);
  w.tr1t = (_Float16*)take((size_t)NIN * NB * 2);
  w.z2h  = (_Float16*)take((size_t)NB * NHID * 2);
  w.tr2t = (_Float16*)take((size_t)NHID * NB * 2);
  w.tr3t = (_Float16*)take((size_t)NOUT * NB * 2);

  float* out1 = (float*)d_out;
  float* out2 = out1 + (size_t)NB * NT * NIN;
  float* out3 = out2 + (size_t)NB * NT * NHID;

  k_init<<<256, 256, 0, stream>>>(w, wi, wh, wo);
  k_iseq<<<(NB * NT) / 16, 256, 0, stream>>>(w, x);
  k_scan<<<16, 256, 0, stream>>>(w, out1, out2, out3, train);
}
